// Attention_89378269430087
// MI455X (gfx1250) — matrix operand order repaired and checked by reading, not compiled, not run on hardware
//
#include <hip/hip_runtime.h>
#include <hip/hip_bf16.h>

// ---------------------------------------------------------------------------
// Types / WMMA fragment helpers (CDNA5 gfx1250, wave32)
// ---------------------------------------------------------------------------
typedef __bf16 bf16_t;
typedef __attribute__((ext_vector_type(16))) __bf16 v16bf;
typedef __attribute__((ext_vector_type(8)))  float  v8f;
typedef __attribute__((ext_vector_type(4)))  __bf16 v4bf;

typedef unsigned int       u32;
typedef unsigned long long u64;
typedef u32 v4u __attribute__((ext_vector_type(4)));
typedef int v8i __attribute__((ext_vector_type(8)));
typedef int v4i __attribute__((ext_vector_type(4)));

union Frag16 { v16bf v; unsigned int u[8]; };

__device__ __forceinline__ v8f vzero8() {
    v8f z = {0.f, 0.f, 0.f, 0.f, 0.f, 0.f, 0.f, 0.f};
    return z;
}

// A-matrix 16x32 bf16 layout (ISA 7.12.2):
//  lanes 0-15: M=lane, VGPR v<4 -> K=2v,2v+1 ; v>=4 -> K=16+2(v-4),+1
//  lanes 16-31: same M, K offset +8
__device__ __forceinline__ void load_a_frag(const bf16_t* base, int stride,
                                            int m0, int k0, Frag16& f, int lane) {
    const int ln = lane & 15, hi = lane >> 4;
    const bf16_t* row = base + (size_t)(m0 + ln) * stride + k0;
#pragma unroll
    for (int v = 0; v < 4; ++v) {
        f.u[v]     = *(const unsigned int*)(row + (2 * v + hi * 8));
        f.u[v + 4] = *(const unsigned int*)(row + (16 + 2 * v + hi * 8));
    }
}

// B-matrix 32x16 bf16 layout: lanes 0-15: N=lane, VGPR v<4 -> K=2v,2v+1 ; v>=4 -> K=16+2(v-4),+1;
// lanes 16-31: K offset +8.  base is row-major [n][k] (i.e. already B^T rows).
__device__ __forceinline__ void load_b_frag(const bf16_t* base, int stride,
                                            int n0, int k0, Frag16& f, int lane) {
    const int ln = lane & 15, hi = lane >> 4;
    const bf16_t* row = base + (size_t)(n0 + ln) * stride + k0;
#pragma unroll
    for (int v = 0; v < 4; ++v) {
        f.u[v]     = *(const unsigned int*)(row + (2 * v + hi * 8));
        f.u[v + 4] = *(const unsigned int*)(row + (16 + 2 * v + hi * 8));
    }
}

__device__ __forceinline__ v8f wmma_bf16(const Frag16& a, const Frag16& b, v8f c) {
    return __builtin_amdgcn_wmma_f32_16x16x32_bf16(
        /*neg_a=*/false, a.v, /*neg_b=*/false, b.v,
        /*c_mod=*/(short)0, c, /*reuse_a=*/false, /*reuse_b=*/false);
}

// ---------------------------------------------------------------------------
// Tensor Data Mover: 2D tile Global -> LDS (ISA cdna5 ch.7/8).
// Descriptor packed per D# spec: group0 = {count, lds_addr, global_addr, type=2},
// group1 = {flags/pad, tensor dims, tile dims, 48-bit row stride}.
// LDS padding done by the TDM itself: pad_interval DWORDs then +1 DWORD pad.
// ---------------------------------------------------------------------------
__device__ __forceinline__ void tdm_load_tile_2d(
    u32 lds_byte_addr, const void* gptr,
    u32 tile_d0,        // elements along contiguous dim
    u32 tile_d1,        // rows
    u64 row_stride,     // elements between rows in global
    u32 pad_interval_enc /* 3 -> every 16 DW, 4 -> every 32 DW; pad = 1 DW */) {
    const u64 ga = (u64)(uintptr_t)gptr;
    v4u g0;
    g0.x = 1u;                                           // count=1 (valid), user mode
    g0.y = lds_byte_addr;                                // lds_addr[31:0]
    g0.z = (u32)(ga & 0xFFFFFFFFu);                      // global_addr[31:0]
    g0.w = (u32)((ga >> 32) & 0x01FFFFFFu) | (2u << 30); // addr[56:32] | type=2

    const u32 dw0 = (1u << 16)                           // data_size = 2 bytes
                  | (1u << 20)                           // pad_enable
                  | (pad_interval_enc << 22);            // pad_amount=0 -> 1 DWORD
    v8i g1;
    g1[0] = (int)dw0;
    g1[1] = (int)((tile_d0 & 0xFFFFu) << 16);            // abar=0 | tensor_dim0[15:0]
    g1[2] = (int)(((tile_d0 >> 16) & 0xFFFFu) | ((tile_d1 & 0xFFFFu) << 16));
    g1[3] = (int)(((tile_d1 >> 16) & 0xFFFFu) | ((tile_d0 & 0xFFFFu) << 16)); // tensor_dim1 hi | tile_dim0
    g1[4] = (int)(tile_d1 & 0xFFFFu);                    // tile_dim1 | tile_dim2=0
    g1[5] = (int)(u32)(row_stride & 0xFFFFFFFFu);        // tensor_dim0_stride[31:0]
    g1[6] = (int)(u32)((row_stride >> 32) & 0xFFFFu);    // stride[47:32] | dim1_stride lo = 0
    g1[7] = 0;
    v4i g2 = {0, 0, 0, 0};
    v4i g3 = {0, 0, 0, 0};
#if defined(__clang_major__) && (__clang_major__ >= 23)
    v8i g4 = {0, 0, 0, 0, 0, 0, 0, 0};
    __builtin_amdgcn_tensor_load_to_lds(g0, g1, g2, g3, g4, 0);
#else
    __builtin_amdgcn_tensor_load_to_lds(g0, g1, g2, g3, 0);
#endif
}

__device__ __forceinline__ u32 lds_addr_of(const void* p) {
    // LDS aperture: flat addr low 32 bits are the LDS byte offset (ISA 10.2)
    return (u32)(uintptr_t)p;
}

// ---------------------------------------------------------------------------
// 16-lane-half xor reductions via ds_swizzle immediates (group-of-32 mode:
// offset = xor_mask<<10 | and_mask 0x1f). Masks <=8 never cross the 16-lane
// halves, matching the WMMA C/D row split (lanes 0-15 row r, 16-31 row r+8).
// NOTE: the swizzle offset must be a frontend constant -> template parameter.
// ---------------------------------------------------------------------------
template <int IMM>
__device__ __forceinline__ float swz(float x) {
    return __builtin_bit_cast(float,
        __builtin_amdgcn_ds_swizzle(__builtin_bit_cast(int, x), IMM));
}
__device__ __forceinline__ float half_max(float v) {
    v = fmaxf(v, swz<0x201F>(v));  // xor 8
    v = fmaxf(v, swz<0x101F>(v));  // xor 4
    v = fmaxf(v, swz<0x081F>(v));  // xor 2
    v = fmaxf(v, swz<0x041F>(v));  // xor 1
    return v;
}
__device__ __forceinline__ float half_sum(float v) {
    v += swz<0x201F>(v);
    v += swz<0x101F>(v);
    v += swz<0x081F>(v);
    v += swz<0x041F>(v);
    return v;
}

// ---------------------------------------------------------------------------
// f32 -> bf16 conversion
// ---------------------------------------------------------------------------
__global__ __launch_bounds__(256) void cvt_f32_bf16_kernel(const float* __restrict__ in,
                                                           bf16_t* __restrict__ out, int n4) {
    int i = blockIdx.x * blockDim.x + threadIdx.x;
    if (i < n4) {
        float4 f = ((const float4*)in)[i];
        v4bf o;
        o.x = (bf16_t)f.x; o.y = (bf16_t)f.y; o.z = (bf16_t)f.z; o.w = (bf16_t)f.w;
        ((v4bf*)out)[i] = o;
    }
}

// ---------------------------------------------------------------------------
// GEMM: Y[r,c] = sum_k X[r,k] * W[c,k] + bias[c]
// X bf16 [R x K], W bf16 [C x K] row-major (=> X @ W^T).
// Block: 256 threads (8 waves), tile 128x128, K-step 32, TDM double buffering.
// Wave grid 2x4; each wave 64 rows x 32 cols = 4x2 C fragments.
// ---------------------------------------------------------------------------
#define LDS_STRIDE 34   // 32 + 2 halfword pad (TDM: pad every 16 DW by 1 DW)

template <bool OUT_F32>
__global__ __launch_bounds__(256) void gemm_bf16_wmma_kernel(
    const bf16_t* __restrict__ X, const bf16_t* __restrict__ W,
    const float* __restrict__ bias, void* __restrict__ Y,
    int R, int C, int K) {
    __shared__ __align__(16) bf16_t Xs[2][128 * LDS_STRIDE];
    __shared__ __align__(16) bf16_t Ws[2][128 * LDS_STRIDE];

    const int tid  = threadIdx.x;
    const int wave = tid >> 5;
    const int lane = tid & 31;
    const int wr = wave >> 2;   // 0..1  (row block of 64)
    const int wc = wave & 3;    // 0..3  (col block of 32)
    const int r0 = blockIdx.y * 128;
    const int c0 = blockIdx.x * 128;

    v8f acc[4][2];
#pragma unroll
    for (int mi = 0; mi < 4; ++mi)
#pragma unroll
        for (int ni = 0; ni < 2; ++ni) acc[mi][ni] = vzero8();

    // Prime buffer 0 via the Tensor Data Mover (one wave issues, TENSORcnt tracks)
    if (wave == 0) {
        tdm_load_tile_2d(lds_addr_of(&Xs[0][0]), X + (size_t)r0 * K, 32, 128, (u64)K, 3);
        tdm_load_tile_2d(lds_addr_of(&Ws[0][0]), W + (size_t)c0 * K, 32, 128, (u64)K, 3);
    }

    const int nk = K >> 5;
    for (int it = 0; it < nk; ++it) {
        const int cur = it & 1;
        __syncthreads();    // all waves done reading buf[cur^1] (previous iter)
        if (wave == 0) {
            if (it + 1 < nk) {
                const int k0n = (it + 1) << 5;
                tdm_load_tile_2d(lds_addr_of(&Xs[cur ^ 1][0]),
                                 X + (size_t)r0 * K + k0n, 32, 128, (u64)K, 3);
                tdm_load_tile_2d(lds_addr_of(&Ws[cur ^ 1][0]),
                                 W + (size_t)c0 * K + k0n, 32, 128, (u64)K, 3);
                __builtin_amdgcn_s_wait_tensorcnt(2);  // 2 oldest (= current buffer) done
            } else {
                __builtin_amdgcn_s_wait_tensorcnt(0);
            }
        }
        __syncthreads();    // current buffer visible to all waves

        Frag16 a[4];
#pragma unroll
        for (int mi = 0; mi < 4; ++mi)
            load_a_frag(&Xs[cur][0], LDS_STRIDE, wr * 64 + mi * 16, 0, a[mi], lane);
        Frag16 b[2];
#pragma unroll
        for (int ni = 0; ni < 2; ++ni)
            load_b_frag(&Ws[cur][0], LDS_STRIDE, wc * 32 + ni * 16, 0, b[ni], lane);

#pragma unroll
        for (int mi = 0; mi < 4; ++mi)
#pragma unroll
            for (int ni = 0; ni < 2; ++ni)
                acc[mi][ni] = wmma_bf16(a[mi], b[ni], acc[mi][ni]);
    }

    // Epilogue: C/D layout -> VGPR v holds row r0+..+v (+8 for upper lanes)
    const int ln = lane & 15, hi = lane >> 4;
#pragma unroll
    for (int ni = 0; ni < 2; ++ni) {
        const int c = c0 + wc * 32 + ni * 16 + ln;
        const float bv = bias[c];
#pragma unroll
        for (int mi = 0; mi < 4; ++mi) {
            const int rbase = r0 + wr * 64 + mi * 16 + hi * 8;
#pragma unroll
            for (int v = 0; v < 8; ++v) {
                const float val = acc[mi][ni][v] + bv;
                const size_t idx = (size_t)(rbase + v) * C + c;
                if (OUT_F32) ((float*)Y)[idx] = val;
                else         ((bf16_t*)Y)[idx] = (bf16_t)val;
            }
        }
    }
}

// ---------------------------------------------------------------------------
// Flash attention (one (b,h) head slice, 64 query rows per workgroup).
// Q,K,V: bf16 [B*S, H*DH] row-major; head h occupies cols [h*64, h*64+64).
// Block: 128 threads (4 waves). Wave w owns query rows [16w,16w+16).
// Q/K tiles staged by TDM; V transposed manually (overlaps the K tensor load).
// Online softmax stats live in registers; ds_swizzle xor reductions over the
// 16-lane halves match the WMMA C/D layout exactly.
// ---------------------------------------------------------------------------
#define AT_STRIDE 66    // 64 + 2 halfword pad (TDM: pad every 32 DW by 1 DW)

__global__ __launch_bounds__(128) void flash_attn_wmma_kernel(
    const bf16_t* __restrict__ Q, const bf16_t* __restrict__ Kp,
    const bf16_t* __restrict__ Vp, bf16_t* __restrict__ O,
    int S /*2048*/, int Hn /*16*/) {
    __shared__ __align__(16) bf16_t Qs[64 * AT_STRIDE];
    __shared__ __align__(16) bf16_t Ks[64 * AT_STRIDE];
    __shared__ __align__(16) bf16_t Vt[64 * AT_STRIDE];   // transposed: Vt[d][k]
    __shared__ __align__(16) bf16_t Ps[4 * 16 * AT_STRIDE];

    const int bh = blockIdx.x;
    const int b = bh / Hn, h = bh % Hn;
    const int m0 = blockIdx.y * 64;
    const int tid = threadIdx.x;
    const int wave = tid >> 5, lane = tid & 31;
    const int ln = lane & 15, hi = lane >> 4;
    const int DM = Hn * 64;   // 1024 row stride

    // --- Q tile (64x64) via TDM, then per-wave A fragments (held all kernel) ---
    if (wave == 0) {
        const bf16_t* Qbase = Q + ((size_t)b * S + m0) * DM + h * 64;
        tdm_load_tile_2d(lds_addr_of(Qs), Qbase, 64, 64, (u64)DM, 4);
        __builtin_amdgcn_s_wait_tensorcnt(0);
    }
    __syncthreads();
    Frag16 qf[2];
#pragma unroll
    for (int kk = 0; kk < 2; ++kk)
        load_a_frag(Qs, AT_STRIDE, wave * 16, kk * 32, qf[kk], lane);

    float mrow[8], lrow[8];
    v8f accO[4];
#pragma unroll
    for (int r = 0; r < 8; ++r) { mrow[r] = -1e30f; lrow[r] = 0.f; }
#pragma unroll
    for (int nb = 0; nb < 4; ++nb) accO[nb] = vzero8();

    for (int n0 = 0; n0 < S; n0 += 64) {
        __syncthreads();   // everyone done with previous Ks/Vt
        // K tile row-major [n][d] via TDM (async while we transpose V below)
        if (wave == 0) {
            const bf16_t* Kbase = Kp + ((size_t)b * S + n0) * DM + h * 64;
            tdm_load_tile_2d(lds_addr_of(Ks), Kbase, 64, 64, (u64)DM, 4);
        }
        // V tile transposed [d][k] so P*V B-fragment k-pairs are contiguous
        {
            const bf16_t* Vbase = Vp + ((size_t)b * S + n0) * DM + h * 64;
#pragma unroll
            for (int itv = 0; itv < 16; ++itv) {
                const int i = tid + itv * 128;          // 2048 bf16-pairs total
                const int kk = i >> 5;                  // key row 0..63
                const int dp = i & 31;                  // d pair 0..31
                union { unsigned int u; bf16_t hval[2]; } pr;
                pr.u = *(const unsigned int*)(Vbase + (size_t)kk * DM + dp * 2);
                Vt[(2 * dp) * AT_STRIDE + kk]     = pr.hval[0];
                Vt[(2 * dp + 1) * AT_STRIDE + kk] = pr.hval[1];
            }
        }
        if (wave == 0) __builtin_amdgcn_s_wait_tensorcnt(0);
        __syncthreads();

        // --- S = Q K^T  (wave: 16 rows x 64 cols) ---
        v8f s[4];
#pragma unroll
        for (int nb = 0; nb < 4; ++nb) s[nb] = vzero8();
#pragma unroll
        for (int kk = 0; kk < 2; ++kk) {
#pragma unroll
            for (int nb = 0; nb < 4; ++nb) {
                Frag16 bk;
                load_b_frag(Ks, AT_STRIDE, nb * 16, kk * 32, bk, lane);
                s[nb] = wmma_bf16(qf[kk], bk, s[nb]);
            }
        }

        // --- online softmax (row stats via swizzle xor reductions) ---
        float alpha[8];
#pragma unroll
        for (int r = 0; r < 8; ++r) {
            float t = s[0][r];
            t = fmaxf(t, s[1][r]); t = fmaxf(t, s[2][r]); t = fmaxf(t, s[3][r]);
            t = half_max(t);
            const float mnew = fmaxf(mrow[r], t);
            alpha[r] = __expf(mrow[r] - mnew);
            mrow[r] = mnew;
        }
#pragma unroll
        for (int nb = 0; nb < 4; ++nb)
#pragma unroll
            for (int r = 0; r < 8; ++r)
                s[nb][r] = __expf(s[nb][r] - mrow[r]);
#pragma unroll
        for (int r = 0; r < 8; ++r) {
            float t = s[0][r] + s[1][r] + s[2][r] + s[3][r];
            t = half_sum(t);
            lrow[r] = lrow[r] * alpha[r] + t;
        }
#pragma unroll
        for (int nb = 0; nb < 4; ++nb)
#pragma unroll
            for (int r = 0; r < 8; ++r)
                accO[nb][r] *= alpha[r];

        // --- P: C/D layout -> A layout through per-wave LDS scratch ---
        bf16_t* Pw = &Ps[wave * 16 * AT_STRIDE];
#pragma unroll
        for (int nb = 0; nb < 4; ++nb)
#pragma unroll
            for (int r = 0; r < 8; ++r)
                Pw[(r + hi * 8) * AT_STRIDE + nb * 16 + ln] = (bf16_t)s[nb][r];
        // wave-local LDS RAW: same-wave DS ops are in-order; make the
        // dependency explicit for HW + compiler (CDNA5 split counters)
        asm volatile("s_wait_dscnt 0" ::: "memory");

        // --- O += P * V ---
#pragma unroll
        for (int kk = 0; kk < 2; ++kk) {
            Frag16 pa;
            load_a_frag(Pw, AT_STRIDE, 0, kk * 32, pa, lane);
#pragma unroll
            for (int nb = 0; nb < 4; ++nb) {
                Frag16 vb;
                load_b_frag(Vt, AT_STRIDE, nb * 16, kk * 32, vb, lane);
                accO[nb] = wmma_bf16(pa, vb, accO[nb]);
            }
        }
    }

    // --- normalize and store merged-head output (bf16 [B*S, H*DH]) ---
    float inv[8];
#pragma unroll
    for (int r = 0; r < 8; ++r) inv[r] = 1.f / lrow[r];
#pragma unroll
    for (int nb = 0; nb < 4; ++nb)
#pragma unroll
        for (int r = 0; r < 8; ++r) {
            const size_t row = (size_t)b * S + m0 + wave * 16 + r + hi * 8;
            O[row * DM + h * 64 + nb * 16 + ln] = (bf16_t)(accO[nb][r] * inv[r]);
        }
}

// ---------------------------------------------------------------------------
// Host launcher
// ---------------------------------------------------------------------------
extern "C" void kernel_launch(void* const* d_in, const int* in_sizes, int n_in,
                              void* d_out, int out_size, void* d_ws, size_t ws_size,
                              hipStream_t stream) {
    const float* queries = (const float*)d_in[0];
    const float* keys    = (const float*)d_in[1];
    const float* values  = (const float*)d_in[2];
    const float* Wq = (const float*)d_in[3];
    const float* bq = (const float*)d_in[4];
    const float* Wk = (const float*)d_in[5];
    const float* bk = (const float*)d_in[6];
    const float* Wv = (const float*)d_in[7];
    const float* bv = (const float*)d_in[8];
    const float* Wo = (const float*)d_in[9];
    const float* bo = (const float*)d_in[10];

    const int Bb = 2, S = 2048, D = 1024, Hn = 16;
    const size_t actN = (size_t)Bb * S * D;   // 4M elements
    const size_t wN   = (size_t)D * D;        // 1M elements

    char* ws = (char*)d_ws;
    bf16_t* qx  = (bf16_t*)ws; ws += actN * sizeof(bf16_t);
    bf16_t* kx  = (bf16_t*)ws; ws += actN * sizeof(bf16_t);
    bf16_t* vx  = (bf16_t*)ws; ws += actN * sizeof(bf16_t);
    bf16_t* wqb = (bf16_t*)ws; ws += wN * sizeof(bf16_t);
    bf16_t* wkb = (bf16_t*)ws; ws += wN * sizeof(bf16_t);
    bf16_t* wvb = (bf16_t*)ws; ws += wN * sizeof(bf16_t);
    bf16_t* wob = (bf16_t*)ws; ws += wN * sizeof(bf16_t);
    bf16_t* Qp  = (bf16_t*)ws; ws += actN * sizeof(bf16_t);
    bf16_t* Kp  = (bf16_t*)ws; ws += actN * sizeof(bf16_t);
    bf16_t* Vp  = (bf16_t*)ws; ws += actN * sizeof(bf16_t);
    bf16_t* Ob  = (bf16_t*)ws; ws += actN * sizeof(bf16_t);

    // 1) convert activations + weights to bf16
    {
        const int na4 = (int)(actN / 4), nw4 = (int)(wN / 4);
        cvt_f32_bf16_kernel<<<(na4 + 255) / 256, 256, 0, stream>>>(queries, qx, na4);
        cvt_f32_bf16_kernel<<<(na4 + 255) / 256, 256, 0, stream>>>(keys,    kx, na4);
        cvt_f32_bf16_kernel<<<(na4 + 255) / 256, 256, 0, stream>>>(values,  vx, na4);
        cvt_f32_bf16_kernel<<<(nw4 + 255) / 256, 256, 0, stream>>>(Wq, wqb, nw4);
        cvt_f32_bf16_kernel<<<(nw4 + 255) / 256, 256, 0, stream>>>(Wk, wkb, nw4);
        cvt_f32_bf16_kernel<<<(nw4 + 255) / 256, 256, 0, stream>>>(Wv, wvb, nw4);
        cvt_f32_bf16_kernel<<<(nw4 + 255) / 256, 256, 0, stream>>>(Wo, wob, nw4);
    }

    const int R = Bb * S;                       // 4096
    const dim3 gemm_grid(D / 128, R / 128);     // (8, 32)

    // 2) Q/K/V projections (bf16 out)
    gemm_bf16_wmma_kernel<false><<<gemm_grid, 256, 0, stream>>>(qx, wqb, bq, Qp, R, D, D);
    gemm_bf16_wmma_kernel<false><<<gemm_grid, 256, 0, stream>>>(kx, wkb, bk, Kp, R, D, D);
    gemm_bf16_wmma_kernel<false><<<gemm_grid, 256, 0, stream>>>(vx, wvb, bv, Vp, R, D, D);

    // 3) flash attention: grid (B*H, M/64)
    flash_attn_wmma_kernel<<<dim3(Bb * Hn, S / 64), 128, 0, stream>>>(Qp, Kp, Vp, Ob, S, Hn);

    // 4) output projection (f32 out + bias)
    gemm_bf16_wmma_kernel<true><<<gemm_grid, 256, 0, stream>>>(Ob, wob, bo, (float*)d_out, R, D, D);
}
